// MetaLearner_61177514164728
// MI455X (gfx1250) — matrix operand order repaired and checked by reading, not compiled, not run on hardware
//
#include <hip/hip_runtime.h>

// ---------------- problem constants (from reference setup_inputs) -----------
#define B_    8
#define C_    1024
#define MID_  256
#define HW_   4096

// ---------------- tiling ----------------------------------------------------
#define NT    64            // spatial columns per workgroup
#define KC    128           // K-chunk of X staged in LDS for GEMM1
#define XSTR  (KC + 8)      // padded LDS column stride (bf16 elems) -> 16B aligned cols
#define HSTR  (MID_ + 8)    // padded LDS column stride for H

typedef __attribute__((ext_vector_type(4)))  float  f4;
typedef __attribute__((ext_vector_type(8)))  float  v8f;
typedef __attribute__((ext_vector_type(2)))  __bf16 v2bf;
typedef __attribute__((ext_vector_type(8)))  __bf16 v8bf;
typedef __attribute__((ext_vector_type(16))) __bf16 v16bf;

static __device__ __forceinline__ v16bf bf_cat(v8bf lo, v8bf hi) {
    return __builtin_shufflevector(lo, hi, 0,1,2,3,4,5,6,7,8,9,10,11,12,13,14,15);
}

// A-fragment (16x32 bf16) from a row-major matrix with K contiguous.
// Layout used here: lane = (l&15) -> row; lanes<16 hold K 0..15, lanes>=16 hold K 16..31 (contiguous).
static __device__ __forceinline__ v16bf load_afrag(const __bf16* rowp, int half) {
    const __bf16* p = rowp + 16 * half;
    return bf_cat(*(const v8bf*)p, *(const v8bf*)(p + 8));
}

// B-fragment (32x16 bf16) from a column-major LDS buffer (K contiguous per column).
// Layout used here: lane = (l&15) -> column; lane half selects K 0..15 / 16..31 (contiguous).
static __device__ __forceinline__ v16bf load_bfrag(const __bf16* colp, int half) {
    const __bf16* p = colp + 16 * half;
    return bf_cat(*(const v8bf*)p, *(const v8bf*)(p + 8));
}

static __device__ __forceinline__ v8f wmma_bf16(v16bf a, v16bf b, v8f c) {
    // (neg_a, A, neg_b, B, c_mod, C, reuse_a, reuse_b)
    return __builtin_amdgcn_wmma_f32_16x16x32_bf16(false, a, false, b, (short)0, c, false, false);
}

// ---------------- weight fp32 -> bf16 pre-conversion (4 MB -> 2 MB, runs once) ----
__global__ __launch_bounds__(256) void convert_weights(
    const float* __restrict__ wd, const float* __restrict__ wu,
    __bf16* __restrict__ wdb, __bf16* __restrict__ wub) {
    const int i = blockIdx.x * 256 + threadIdx.x;   // grid sized exactly to 2*MID_*C_
    wdb[i] = (__bf16)wd[i];
    wub[i] = (__bf16)wu[i];
}

// ---------------- fused  down-proj -> ReLU -> up-proj -> +bias -> +residual ------
__global__ __launch_bounds__(256) void moe_fused(
    const float*  __restrict__ x,
    const __bf16* __restrict__ wd,
    const float*  __restrict__ bd,
    const __bf16* __restrict__ wu,
    const float*  __restrict__ bu,
    const int*    __restrict__ cond,
    float*        __restrict__ out)
{
    __shared__ __align__(16) __bf16 Xs[NT * XSTR];   // X chunk, column-major bf16 (17.0 KB)
    __shared__ __align__(16) __bf16 Hs[NT * HSTR];   // H tile,  column-major bf16 (33.0 KB)

    const int b    = blockIdx.y;
    const int col0 = blockIdx.x * NT;
    const int e    = cond[b];

    const __bf16* Wd_e = wd + (size_t)e * MID_ * C_;
    const __bf16* Wu_e = wu + (size_t)e * C_ * MID_;
    const float*  bd_e = bd + e * MID_;
    const float*  bu_e = bu + e * C_;
    const float*  xb   = x   + (size_t)b * C_ * HW_;
    float*        yb   = out + (size_t)b * C_ * HW_;

    const int tid  = threadIdx.x;
    const int wave = tid >> 5;       // 8 waves
    const int lane = tid & 31;
    const int half = lane >> 4;
    const int l16  = lane & 15;

    // ---------------- GEMM1: H = relu(Wd_e * X + bd), M=256 K=1024 N=64 ----------
    // Wave w owns MID rows [32w, 32w+32) -> 2 row-tiles x 4 col-tiles of f32 acc.
    v8f acc[2][4] = {};

    for (int kk = 0; kk < C_; kk += KC) {
        __syncthreads();   // previous chunk's readers done before Xs overwrite
        // Stage X[kk..kk+KC, col0..col0+63]: fp32 global (coalesced 256B rows) -> bf16 LDS
        // col-major. Two consecutive K rows per thread so each (k,k+1) bf16 pair is one
        // packed 4B ds_store_b32 instead of two ds_store_b16.
        #pragma unroll
        for (int it = 0; it < KC / 32; ++it) {         // 4 iters, 2 x float4 each
            const int r  = it * 32 + (tid >> 4) * 2;   // even row in [0, KC)
            const int c4 = (tid & 15) * 4;
            const f4 v0 = *(const f4*)&xb[(size_t)(kk + r)     * HW_ + col0 + c4];
            const f4 v1 = *(const f4*)&xb[(size_t)(kk + r + 1) * HW_ + col0 + c4];
            #pragma unroll
            for (int j = 0; j < 4; ++j) {
                v2bf p;
                p[0] = (__bf16)v0[j];
                p[1] = (__bf16)v1[j];
                *(v2bf*)&Xs[(c4 + j) * XSTR + r] = p;  // 4B-aligned packed store
            }
        }
        __syncthreads();

        #pragma unroll
        for (int ks = 0; ks < KC; ks += 32) {
            v16bf afrag[2];
            #pragma unroll
            for (int rt = 0; rt < 2; ++rt) {
                const int mrow = 32 * wave + 16 * rt + l16;
                afrag[rt] = load_afrag(Wd_e + (size_t)mrow * C_ + kk + ks, half);
            }
            #pragma unroll
            for (int nc = 0; nc < 4; ++nc) {
                const v16bf bfrag = load_bfrag(&Xs[(16 * nc + l16) * XSTR + ks], half);
                acc[0][nc] = wmma_bf16(afrag[0], bfrag, acc[0][nc]);
                acc[1][nc] = wmma_bf16(afrag[1], bfrag, acc[1][nc]);
            }
        }
    }

    // Epilogue 1: +bd, ReLU, bf16, store to Hs column-major (B-fragment-ready).
    // C/D layout: VGPR r, lane -> (row = base + 8*half + r, col = 16*nc + l16).
    #pragma unroll
    for (int rt = 0; rt < 2; ++rt) {
        const int mb = 32 * wave + 16 * rt + 8 * half;
        const f4 b0 = *(const f4*)&bd_e[mb];
        const f4 b1 = *(const f4*)&bd_e[mb + 4];
        #pragma unroll
        for (int nc = 0; nc < 4; ++nc) {
            v8bf h;
            #pragma unroll
            for (int r = 0; r < 8; ++r) {
                float v = acc[rt][nc][r] + (r < 4 ? b0[r] : b1[r - 4]);
                v = v > 0.f ? v : 0.f;
                h[r] = (__bf16)v;
            }
            *(v8bf*)&Hs[(16 * nc + l16) * HSTR + mb] = h;   // 16B DS store
        }
    }
    __syncthreads();

    // ---------------- GEMM2: Y = Wu_e * H + bu + X, M=1024 K=256 N=64 ------------
    // Wave w sweeps row-tiles {w, w+8, ..., w+56}; acc stays at 4 tiles (32 VGPRs).
    #pragma unroll 1
    for (int i = 0; i < 8; ++i) {
        const int rt = wave + 8 * i;
        v8f acc2[4] = {};
        #pragma unroll
        for (int ks = 0; ks < MID_; ks += 32) {
            const int crow = 16 * rt + l16;
            const v16bf a = load_afrag(Wu_e + (size_t)crow * MID_ + ks, half);
            #pragma unroll
            for (int nc = 0; nc < 4; ++nc) {
                const v16bf bfrag = load_bfrag(&Hs[(16 * nc + l16) * HSTR + ks], half);
                acc2[nc] = wmma_bf16(a, bfrag, acc2[nc]);
            }
        }
        const int cb = 16 * rt + 8 * half;
        const f4 u0 = *(const f4*)&bu_e[cb];
        const f4 u1 = *(const f4*)&bu_e[cb + 4];
        #pragma unroll
        for (int nc = 0; nc < 4; ++nc) {
            const int col = col0 + 16 * nc + l16;
            #pragma unroll
            for (int r = 0; r < 8; ++r) {
                const size_t off = (size_t)(cb + r) * HW_ + col;
                yb[off] = acc2[nc][r] + (r < 4 ? u0[r] : u1[r - 4]) + xb[off];
            }
        }
    }
}

extern "C" void kernel_launch(void* const* d_in, const int* in_sizes, int n_in,
                              void* d_out, int out_size, void* d_ws, size_t ws_size,
                              hipStream_t stream) {
    (void)in_sizes; (void)n_in; (void)out_size; (void)ws_size;
    const float* x    = (const float*)d_in[0];
    const float* Wd   = (const float*)d_in[1];
    const float* bd   = (const float*)d_in[2];
    const float* Wu   = (const float*)d_in[3];
    const float* bu   = (const float*)d_in[4];
    const int*   cond = (const int*)  d_in[5];
    float* out = (float*)d_out;

    __bf16* wdb = (__bf16*)d_ws;                       // 2*256*1024 bf16 = 1 MB
    __bf16* wub = wdb + (size_t)2 * MID_ * C_;         // 2*1024*256 bf16 = 1 MB

    convert_weights<<<(2 * MID_ * C_) / 256, 256, 0, stream>>>(Wd, Wu, wdb, wub);

    dim3 grid(HW_ / NT, B_);                           // 64 col-tiles x 8 samples = 512 WGs
    moe_fused<<<grid, 256, 0, stream>>>(x, wdb, bd, wub, bu, cond, out);
}
